// VisionTransformer_80977313399441
// MI455X (gfx1250) — compile-verified, matrix operand order checked by reading, not run on hardware
//
#include <hip/hip_runtime.h>
#include <hip/hip_bf16.h>
#include <cstring>
#include <cstdint>

// ---------------------------------------------------------------------------
// ViT-S/16 forward for MI455X (gfx1250), wave32 + WMMA f32=f16*f16+f32.
// Residual stream fp32; all GEMM operands f16; fp32 accumulation in WMMA.
// Seq padded to SP=224 and Wcls padded to 1024 so EVERY GEMM is edge-free.
// K-loop: register prefetch + double-buffered LDS (one barrier per k-step)
// + L2 prefetch two steps ahead (global_prefetch_b8).
// ---------------------------------------------------------------------------

typedef __attribute__((ext_vector_type(16))) _Float16 v16h;
typedef __attribute__((ext_vector_type(8)))  float    v8f;
typedef __attribute__((ext_vector_type(8)))  _Float16 h8;

namespace vitc {
constexpr int Bb   = 64;
constexpr int Cc   = 3;
constexpr int Hh   = 224;
constexpr int Ww   = 224;
constexpr int NP   = 14;
constexpr int PS   = 16;
constexpr int IND  = Cc * PS * PS;   // 768
constexpr int D    = 384;
constexpr int L    = 12;
constexpr int NH   = 6;
constexpr int HS   = 64;
constexpr int S    = NP * NP + 1;    // 197
constexpr int SP   = 224;            // S padded to multiple of 32
constexpr int FF   = 4 * D;          // 1536
constexpr int NC   = 1000;
constexpr int NCP  = 1024;           // NC padded to multiple of 32
constexpr int TOK  = Bb * SP;        // 14336 padded token rows
}

// ---------------------------------------------------------------------------
// Batched WMMA GEMM (edge-free):  D = alpha*(A@B) + bias + residual, act opt.
// A: f16 [M,K] row-major (lda).  B: f16 [K,N] row-major (ldb) or, if TRANSB,
// stored [N,K] row-major and consumed as B[k][n] = Bm[n*ldb + k].
// Requires: M,N,K multiples of 32; lda/ldb/ldd multiples of 8 halves;
// base pointers 16B-aligned.
// ---------------------------------------------------------------------------
struct GemmP {
  const _Float16* A;   long long sAb, sAh; int lda;
  const _Float16* Bm;  long long sBb, sBh; int ldb;
  float*          C;   long long sCb, sCh;
  _Float16*       C16; long long s16b, s16h;
  int ldd;
  const float* bias;   long long sbias;
  const float* res;    long long sresb, sresh;
  int M, N, K;
  float alpha;
  int act;       // 0 = none, 1 = leaky-relu(0.01)
  int hcount;
};

#define BLK 32
#define LDSS 40             // LDS row stride in halves: 80B rows, 16B-aligned
#define TILEH (BLK * LDSS)  // halves per tile buffer

template<bool TRANSB>
__global__ __launch_bounds__(128)
void gemm_wmma_kernel(GemmP p) {
  __shared__ _Float16 As [2 * TILEH];   // [buf][m][k]
  __shared__ _Float16 BsT[2 * TILEH];   // [buf][n][k]  (transposed tile)

  const int tid  = threadIdx.x;
  const int lane = tid & 31;
  const int wid  = tid >> 5;            // 0..3 -> 2x2 wave grid of 16x16 tiles
  const int wrow = (wid >> 1) << 4;
  const int wcol = (wid & 1) << 4;

  const long long bm = (long long)blockIdx.y * BLK;
  const long long bn = (long long)blockIdx.x * BLK;

  const int z    = blockIdx.z;
  const int bidx = z / p.hcount;
  const int hidx = z % p.hcount;

  const _Float16* Aof = p.A  + bidx * p.sAb + hidx * p.sAh;
  const _Float16* Bof = p.Bm + bidx * p.sBb + hidx * p.sBh;
  float*     C   = p.C   ? p.C   + bidx * p.sCb  + hidx * p.sCh  : nullptr;
  _Float16*  C16 = p.C16 ? p.C16 + bidx * p.s16b + hidx * p.s16h : nullptr;
  const float* bias = p.bias ? p.bias + hidx * p.sbias : nullptr;
  const float* res  = p.res  ? p.res + bidx * p.sresb + hidx * p.sresh : nullptr;

  v8f acc = {};

  const int lr = tid >> 2;        // 0..31
  const int lc = (tid & 3) << 3;  // 0,8,16,24

  // per-thread global source pointers (advance along K)
  const _Float16* aPtr = Aof + (bm + lr) * p.lda + lc;
  const _Float16* bPtr = TRANSB ? Bof + (bn + lr) * p.ldb + lc
                                : Bof + (long long)lr * p.ldb + bn + lc;

  // fragment LDS indices (ISA 7.12.2 wave32 layouts)
  const int m    = wrow + (lane & 15);
  const int n    = wcol + (lane & 15);
  const int koff = (lane >> 4) << 3;   // 0 or 8

  // ---- prologue: fetch tile 0 and commit to buffer 0 ----
  h8 aReg = *(const h8*)aPtr;
  h8 bReg = *(const h8*)bPtr;
  *(h8*)&As[lr * LDSS + lc] = aReg;
  if constexpr (TRANSB) {
    *(h8*)&BsT[lr * LDSS + lc] = bReg;
  } else {
    #pragma unroll
    for (int i = 0; i < 8; ++i) BsT[(lc + i) * LDSS + lr] = bReg[i];
  }

  int buf = 0;
  for (int k0 = 0; k0 < p.K; k0 += BLK) {
    __syncthreads();   // stores to buf visible; buf^1 free to overwrite

    const bool more = (k0 + BLK) < p.K;
    // ---- issue next tile's global loads (overlap with WMMA below) ----
    if (more) {
      aReg = *(const h8*)(aPtr + (k0 + BLK));
      bReg = TRANSB ? *(const h8*)(bPtr + (k0 + BLK))
                    : *(const h8*)(bPtr + (long long)(k0 + BLK) * p.ldb);
    }
    // ---- L2 prefetch two steps ahead (global_prefetch_b8) ----
    if (k0 + 2 * BLK < p.K) {
      __builtin_prefetch(aPtr + (k0 + 2 * BLK), 0, 3);
      __builtin_prefetch(TRANSB ? bPtr + (k0 + 2 * BLK)
                                : bPtr + (long long)(k0 + 2 * BLK) * p.ldb, 0, 3);
    }

    // ---- compute from current buffer ----
    const int lb = buf * TILEH;
    const h8 a0 = *(const h8*)&As [lb + m * LDSS + koff];
    const h8 a1 = *(const h8*)&As [lb + m * LDSS + 16 + koff];
    const h8 b0 = *(const h8*)&BsT[lb + n * LDSS + koff];
    const h8 b1 = *(const h8*)&BsT[lb + n * LDSS + 16 + koff];
    const v16h af = __builtin_shufflevector(a0, a1,
        0,1,2,3,4,5,6,7,8,9,10,11,12,13,14,15);
    const v16h bf = __builtin_shufflevector(b0, b1,
        0,1,2,3,4,5,6,7,8,9,10,11,12,13,14,15);
    acc = __builtin_amdgcn_wmma_f32_16x16x32_f16(
        false, af, false, bf, (short)0, acc, false, false);

    // ---- commit prefetched tile into the other buffer ----
    if (more) {
      const int ob = (buf ^ 1) * TILEH;
      *(h8*)&As[ob + lr * LDSS + lc] = aReg;
      if constexpr (TRANSB) {
        *(h8*)&BsT[ob + lr * LDSS + lc] = bReg;
      } else {
        #pragma unroll
        for (int i = 0; i < 8; ++i) BsT[ob + (lc + i) * LDSS + lr] = bReg[i];
      }
    }
    buf ^= 1;
  }

  // ---- epilogue: D layout  M = v + 8*(lane>>4), N = lane&15 ----
  const int half = lane >> 4;
  const long long col = bn + wcol + (lane & 15);
  #pragma unroll
  for (int v = 0; v < 8; ++v) {
    const long long row = bm + wrow + v + 8 * half;
    float f = p.alpha * acc[v];
    if (bias) f += bias[col];
    if (res)  f += res[row * p.ldd + col];
    if (p.act) f = (f > 0.0f) ? f : 0.01f * f;
    if (C)   C[row * p.ldd + col] = f;
    if (C16) C16[row * p.ldd + col] = (_Float16)f;
  }
}

// ---------------------------------------------------------------------------
// Elementwise / reduction kernels
// ---------------------------------------------------------------------------
__global__ void cvt_f32_f16_kernel(const float* __restrict__ src,
                                   _Float16* __restrict__ dst, long long n) {
  long long i = (long long)blockIdx.x * blockDim.x + threadIdx.x;
  if (i < n) dst[i] = (_Float16)src[i];
}

// Wcls [D, NC] -> f16 [D, NCP] with zero padding of the last columns
__global__ void cvt_pad_wcls_kernel(const float* __restrict__ src,
                                    _Float16* __restrict__ dst) {
  using namespace vitc;
  const int i = blockIdx.x * blockDim.x + threadIdx.x;   // over D*NCP
  if (i < D * NCP) {
    const int r = i >> 10;          // / NCP
    const int c = i & (NCP - 1);    // % NCP
    dst[i] = (c < NC) ? (_Float16)src[r * NC + c] : (_Float16)0.0f;
  }
}

__global__ void patchify_kernel(const float* __restrict__ x,
                                _Float16* __restrict__ P) {
  using namespace vitc;
  long long idx = (long long)blockIdx.x * blockDim.x + threadIdx.x;
  const long long total = (long long)Bb * NP * NP * IND;
  if (idx >= total) return;
  const int inner = (int)(idx % IND);
  long long rest  = idx / IND;
  const int patch = (int)(rest % (NP * NP));
  const int b     = (int)(rest / (NP * NP));
  const int c  = inner >> 8;          // /256
  const int pr = (inner >> 4) & 15;
  const int pc = inner & 15;
  const int pi = patch / NP, pj = patch % NP;
  const int hh = pi * PS + pr, ww = pj * PS + pc;
  P[idx] = (_Float16)x[(((long long)b * Cc + c) * Hh + hh) * Ww + ww];
}

__global__ void assemble_kernel(const float* __restrict__ e,
                                const float* __restrict__ cls,
                                const float* __restrict__ pos,
                                float* __restrict__ t) {
  using namespace vitc;
  long long idx = (long long)blockIdx.x * blockDim.x + threadIdx.x;
  const long long total = (long long)TOK * D;
  if (idx >= total) return;
  const int d = (int)(idx % D);
  long long rest = idx / D;
  const int s = (int)(rest % SP);
  const int b = (int)(rest / SP);
  float v;
  if (s == 0)       v = cls[d] + pos[d];
  else if (s < S)   v = e[((long long)b * (NP * NP) + s - 1) * D + d]
                        + pos[(long long)s * D + d];
  else              v = 0.0f;
  t[idx] = v;
}

__global__ __launch_bounds__(128)
void layernorm_kernel(const float* __restrict__ x, const float* __restrict__ g,
                      const float* __restrict__ bsh, _Float16* __restrict__ out) {
  using namespace vitc;
  const int r = blockIdx.x;              // 0 .. TOK-1
  const int s = r % SP;
  const int tid = threadIdx.x;
  __shared__ float red[128];
  const float* row = x + (long long)r * D;
  _Float16* orow = out + (long long)r * D;
  if (s >= S) {                          // zero padding rows
    for (int i = tid; i < D; i += 128) orow[i] = (_Float16)0.0f;
    return;
  }
  float sum = 0.0f, sq = 0.0f;
  for (int i = tid; i < D; i += 128) { float v = row[i]; sum += v; sq += v * v; }
  red[tid] = sum; __syncthreads();
  for (int k = 64; k > 0; k >>= 1) { if (tid < k) red[tid] += red[tid + k]; __syncthreads(); }
  const float mean = red[0] / D;
  __syncthreads();
  red[tid] = sq; __syncthreads();
  for (int k = 64; k > 0; k >>= 1) { if (tid < k) red[tid] += red[tid + k]; __syncthreads(); }
  const float var = red[0] / D - mean * mean;
  const float inv = rsqrtf(var + 1e-5f);
  for (int i = tid; i < D; i += 128)
    orow[i] = (_Float16)((row[i] - mean) * inv * g[i] + bsh[i]);
}

__global__ __launch_bounds__(256)
void softmax_kernel(const float* __restrict__ scores,
                    _Float16* __restrict__ probs) {
  using namespace vitc;
  const long long r = blockIdx.x;        // 0 .. B*NH*SP-1
  const int s = (int)(r % SP);
  const float* row = scores + r * SP;
  _Float16* prow = probs + r * SP;
  const int tid = threadIdx.x;
  __shared__ float red[256];
  if (s >= S) {                          // zero padding rows (A of P@V GEMM)
    if (tid < SP) prow[tid] = (_Float16)0.0f;
    return;
  }
  const float v = (tid < S) ? row[tid] : -3.0e38f;
  red[tid] = v; __syncthreads();
  for (int k = 128; k > 0; k >>= 1) { if (tid < k) red[tid] = fmaxf(red[tid], red[tid + k]); __syncthreads(); }
  const float m = red[0];
  __syncthreads();
  const float e = (tid < S) ? __expf(v - m) : 0.0f;
  red[tid] = e; __syncthreads();
  for (int k = 128; k > 0; k >>= 1) { if (tid < k) red[tid] += red[tid + k]; __syncthreads(); }
  const float inv = 1.0f / red[0];
  if (tid < SP) prow[tid] = (_Float16)(e * inv);   // cols >= S get exact 0
}

__global__ void extract_cls_kernel(const float* __restrict__ t,
                                   _Float16* __restrict__ cls16) {
  using namespace vitc;
  const int i = blockIdx.x * blockDim.x + threadIdx.x;
  if (i < Bb * D) {
    const int b = i / D, d = i % D;
    cls16[i] = (_Float16)t[(long long)b * SP * D + d];
  }
}

// out[b, c] = outpad[b, c] + bcls[c]   (drops the NCP padding)
__global__ void copy_head_kernel(const float* __restrict__ outpad,
                                 const float* __restrict__ bcls,
                                 float* __restrict__ out) {
  using namespace vitc;
  const int i = blockIdx.x * blockDim.x + threadIdx.x;
  if (i < Bb * NC) {
    const int b = i / NC, c = i % NC;
    out[i] = outpad[(long long)b * NCP + c] + bcls[c];
  }
}

// ---------------------------------------------------------------------------
// Host-side orchestration
// ---------------------------------------------------------------------------
static inline GemmP gp0() {
  GemmP p; std::memset(&p, 0, sizeof(p));
  p.alpha = 1.0f; p.hcount = 1;
  return p;
}

extern "C" void kernel_launch(void* const* d_in, const int* in_sizes, int n_in,
                              void* d_out, int out_size, void* d_ws, size_t ws_size,
                              hipStream_t stream) {
  using namespace vitc;
  (void)in_sizes; (void)n_in; (void)out_size; (void)ws_size;

  const float* x      = (const float*)d_in[0];
  const float* Wmap   = (const float*)d_in[1];
  const float* bmap   = (const float*)d_in[2];
  const float* cls    = (const float*)d_in[3];
  const float* pos    = (const float*)d_in[4];
  const float* ln1_g  = (const float*)d_in[5];
  const float* ln1_b  = (const float*)d_in[6];
  const float* Wq     = (const float*)d_in[7];
  const float* bq     = (const float*)d_in[8];
  const float* Wk     = (const float*)d_in[9];
  const float* bk     = (const float*)d_in[10];
  const float* Wv     = (const float*)d_in[11];
  const float* bv     = (const float*)d_in[12];
  const float* Wproj  = (const float*)d_in[13];
  const float* bproj  = (const float*)d_in[14];
  const float* ln2_g  = (const float*)d_in[15];
  const float* ln2_b  = (const float*)d_in[16];
  const float* Wff1   = (const float*)d_in[17];
  const float* bff1   = (const float*)d_in[18];
  const float* Wff2   = (const float*)d_in[19];
  const float* bff2   = (const float*)d_in[20];
  const float* Wcls   = (const float*)d_in[21];
  const float* bcls   = (const float*)d_in[22];
  float* out = (float*)d_out;

  // ---- workspace carving ----
  size_t off = 0;
  auto alloc = [&](size_t bytes) -> char* {
    size_t o = off; off = (off + bytes + 255) & ~(size_t)255;
    return (char*)d_ws + o;
  };
  const long long nWmap = (long long)IND * D;
  const long long nWqkv = (long long)L * NH * D * HS;
  const long long nWpr  = (long long)L * D * D;
  const long long nWf1  = (long long)L * D * FF;
  const long long nWf2  = (long long)L * FF * D;

  _Float16* wmap16 = (_Float16*)alloc(nWmap * 2);
  _Float16* wq16   = (_Float16*)alloc(nWqkv * 2);
  _Float16* wk16   = (_Float16*)alloc(nWqkv * 2);
  _Float16* wv16   = (_Float16*)alloc(nWqkv * 2);
  _Float16* wpr16  = (_Float16*)alloc(nWpr * 2);
  _Float16* wf1_16 = (_Float16*)alloc(nWf1 * 2);
  _Float16* wf2_16 = (_Float16*)alloc(nWf2 * 2);
  _Float16* wcl16  = (_Float16*)alloc((long long)D * NCP * 2);

  _Float16* P16    = (_Float16*)alloc((long long)Bb * NP * NP * IND * 2);
  float*    e32    = (float*)   alloc((long long)Bb * NP * NP * D * 4);
  float*    t32    = (float*)   alloc((long long)TOK * D * 4);
  _Float16* hn16   = (_Float16*)alloc((long long)TOK * D * 2);
  _Float16* q16    = (_Float16*)alloc((long long)NH * TOK * HS * 2);
  _Float16* k16    = (_Float16*)alloc((long long)NH * TOK * HS * 2);
  _Float16* v16    = (_Float16*)alloc((long long)NH * TOK * HS * 2);
  float*    sc32   = (float*)   alloc((long long)Bb * NH * SP * SP * 4);
  _Float16* pr16   = (_Float16*)alloc((long long)Bb * NH * SP * SP * 2);
  _Float16* o16    = (_Float16*)alloc((long long)TOK * D * 2);
  _Float16* ff16   = (_Float16*)alloc((long long)TOK * FF * 2);
  _Float16* cls16  = (_Float16*)alloc((long long)Bb * D * 2);
  float*    outpad = (float*)   alloc((long long)Bb * NCP * 4);

  auto cvt = [&](const float* s, _Float16* d, long long n) {
    cvt_f32_f16_kernel<<<dim3((unsigned)((n + 255) / 256)), 256, 0, stream>>>(s, d, n);
  };
  auto launch_gemm = [&](const GemmP& p, int zc, bool transB) {
    dim3 grid(p.N / BLK, p.M / BLK, zc);
    if (transB) gemm_wmma_kernel<true ><<<grid, 128, 0, stream>>>(p);
    else        gemm_wmma_kernel<false><<<grid, 128, 0, stream>>>(p);
  };

  // ---- weights -> f16 (recomputed each call; deterministic) ----
  cvt(Wmap, wmap16, nWmap);
  cvt(Wq,   wq16,   nWqkv);
  cvt(Wk,   wk16,   nWqkv);
  cvt(Wv,   wv16,   nWqkv);
  cvt(Wproj,wpr16,  nWpr);
  cvt(Wff1, wf1_16, nWf1);
  cvt(Wff2, wf2_16, nWf2);
  cvt_pad_wcls_kernel<<<dim3((D * NCP + 255) / 256), 256, 0, stream>>>(Wcls, wcl16);

  // ---- patch embedding ----
  {
    const long long np = (long long)Bb * NP * NP * IND;
    patchify_kernel<<<dim3((unsigned)((np + 255) / 256)), 256, 0, stream>>>(x, P16);
    GemmP p = gp0();
    p.A = P16; p.lda = IND; p.Bm = wmap16; p.ldb = D;
    p.C = e32; p.ldd = D; p.bias = bmap;
    p.M = Bb * NP * NP; p.N = D; p.K = IND;
    launch_gemm(p, 1, false);
    const long long na = (long long)TOK * D;
    assemble_kernel<<<dim3((unsigned)((na + 255) / 256)), 256, 0, stream>>>(e32, cls, pos, t32);
  }

  const long long headStride = (long long)TOK * HS;   // q/k/v per-head slab

  for (int l = 0; l < L; ++l) {
    // LN1
    layernorm_kernel<<<dim3(TOK), 128, 0, stream>>>(
        t32, ln1_g + (long long)l * D, ln1_b + (long long)l * D, hn16);

    // Q, K, V  (batched over heads)
    for (int which = 0; which < 3; ++which) {
      GemmP p = gp0();
      p.A = hn16; p.lda = D;
      p.Bm  = (which == 0 ? wq16 : which == 1 ? wk16 : wv16) + (long long)l * NH * D * HS;
      p.sBh = (long long)D * HS; p.ldb = HS;
      p.C16 = (which == 0 ? q16 : which == 1 ? k16 : v16);
      p.s16h = headStride; p.ldd = HS;
      p.bias = (which == 0 ? bq : which == 1 ? bk : bv) + (long long)l * NH * HS;
      p.sbias = HS;
      p.M = TOK; p.N = HS; p.K = D; p.hcount = NH;
      launch_gemm(p, NH, false);
    }

    // scores = (Q @ K^T) / sqrt(HS)   per (b,h)  [M=N=SP, K=HS]
    {
      GemmP p = gp0();
      p.A = q16;  p.sAb = (long long)SP * HS; p.sAh = headStride; p.lda = HS;
      p.Bm = k16; p.sBb = (long long)SP * HS; p.sBh = headStride; p.ldb = HS;
      p.C = sc32; p.sCb = (long long)NH * SP * SP; p.sCh = (long long)SP * SP;
      p.ldd = SP;
      p.M = SP; p.N = SP; p.K = HS;
      p.alpha = 0.125f; p.hcount = NH;
      launch_gemm(p, Bb * NH, true);
    }

    // softmax (masked to S; zero-fills padded rows and cols)
    softmax_kernel<<<dim3(Bb * NH * SP), 256, 0, stream>>>(sc32, pr16);

    // O = P @ V   per (b,h)  [K=SP exact since padded probs are zero]
    {
      GemmP p = gp0();
      p.A = pr16; p.sAb = (long long)NH * SP * SP; p.sAh = (long long)SP * SP; p.lda = SP;
      p.Bm = v16; p.sBb = (long long)SP * HS; p.sBh = headStride; p.ldb = HS;
      p.C16 = o16; p.s16b = (long long)SP * D; p.s16h = HS; p.ldd = D;
      p.M = SP; p.N = HS; p.K = SP; p.hcount = NH;
      launch_gemm(p, Bb * NH, false);
    }

    // t = t + O @ Wproj + bproj
    {
      GemmP p = gp0();
      p.A = o16; p.lda = D;
      p.Bm = wpr16 + (long long)l * D * D; p.ldb = D;
      p.C = t32; p.ldd = D;
      p.bias = bproj + (long long)l * D;
      p.res = t32;
      p.M = TOK; p.N = D; p.K = D;
      launch_gemm(p, 1, false);
    }

    // LN2
    layernorm_kernel<<<dim3(TOK), 128, 0, stream>>>(
        t32, ln2_g + (long long)l * D, ln2_b + (long long)l * D, hn16);

    // FF1 + leaky relu
    {
      GemmP p = gp0();
      p.A = hn16; p.lda = D;
      p.Bm = wf1_16 + (long long)l * D * FF; p.ldb = FF;
      p.C16 = ff16; p.ldd = FF;
      p.bias = bff1 + (long long)l * FF;
      p.act = 1;
      p.M = TOK; p.N = FF; p.K = D;
      launch_gemm(p, 1, false);
    }

    // t = t + FF @ Wff2 + bff2
    {
      GemmP p = gp0();
      p.A = ff16; p.lda = FF;
      p.Bm = wf2_16 + (long long)l * FF * D; p.ldb = D;
      p.C = t32; p.ldd = D;
      p.bias = bff2 + (long long)l * D;
      p.res = t32;
      p.M = TOK; p.N = D; p.K = FF;
      launch_gemm(p, 1, false);
    }
  }

  // classifier head on cls token (padded to NCP=1024 -> edge-free GEMM)
  extract_cls_kernel<<<dim3((Bb * D + 255) / 256), 256, 0, stream>>>(t32, cls16);
  {
    GemmP p = gp0();
    p.A = cls16; p.lda = D;
    p.Bm = wcl16; p.ldb = NCP;
    p.C = outpad; p.ldd = NCP;
    p.M = Bb; p.N = NCP; p.K = D;
    launch_gemm(p, 1, false);
  }
  copy_head_kernel<<<dim3((Bb * NC + 255) / 256), 256, 0, stream>>>(outpad, bcls, out);
}
